// FNOProjection_62706522521940
// MI455X (gfx1250) — matrix operand order repaired and checked by reading, not compiled, not run on hardware
//
#include <hip/hip_runtime.h>
#include <hip/hip_bf16.h>
#include <hip/hip_fp16.h>

// ---------------------------------------------------------------------------
// FNO forward for MI455X (gfx1250, wave32, WMMA + TDM).
// Spectral conv realized as dense GEMMs vs precomputed DFT bases stored in
// WMMA-fragment order. f16 operands, f32 accumulation. Tile staging uses the
// Tensor Data Mover (tensor_load_to_lds) with double-buffered LDS so the DMA
// overlaps the WMMA stream; TENSORcnt pipelined via s_wait_tensorcnt.
// ---------------------------------------------------------------------------

typedef __attribute__((ext_vector_type(16))) _Float16 v16h;
typedef __attribute__((ext_vector_type(8)))  float    v8f;
typedef __attribute__((ext_vector_type(4)))  uint32_t v4u;
typedef __attribute__((ext_vector_type(8)))  int      v8i_;
typedef __attribute__((ext_vector_type(4)))  int      v4i_;

#define BATCH 32
#define LSEQ  4096
#define HCH   64
#define MMODE 64
#define K2    128   // 2*MMODE (real | imag)
#define KCHUNK 256  // K staged per stage in fwd_dft (8 WMMA steps)
#define TWO_PI 6.28318530717958647692f

#if __has_builtin(__builtin_amdgcn_tensor_load_to_lds)
#define HAVE_TDM 1
#else
#define HAVE_TDM 0
#endif

union Frag { v16h v; float4 f4[2]; };

__device__ __forceinline__ float gelu_tanh(float x) {
    const float c = 0.7978845608028654f;  // sqrt(2/pi)
    float t = tanhf(c * (x + 0.044715f * x * x * x));
    return 0.5f * x * (1.0f + t);
}

#if HAVE_TDM
// Issue a 2D TDM tile load (group1 supplies dims/strides; group0 built here).
// Descriptor words forced to SGPRs via readfirstlane (wave-uniform by
// construction). D# group0: count=1 | lds_addr | 57-bit global addr | type=2.
__device__ __forceinline__ void tdm_issue_2d(uint32_t ldsoff, const void* gptr,
                                             v8i_ g1) {
    uint64_t ga = (uint64_t)(uintptr_t)gptr;
    v4u g0;
    g0.x = 1u;
    g0.y = (uint32_t)__builtin_amdgcn_readfirstlane((int)ldsoff);
    g0.z = (uint32_t)__builtin_amdgcn_readfirstlane((int)(uint32_t)ga);
    g0.w = (uint32_t)__builtin_amdgcn_readfirstlane(
               (int)((uint32_t)(ga >> 32) | (2u << 30)));
    const v4i_ z4 = {0, 0, 0, 0};
#if __clang_major__ >= 23
    const v8i_ z8 = {0, 0, 0, 0, 0, 0, 0, 0};
    __builtin_amdgcn_tensor_load_to_lds(g0, g1, z4, z4, z8, 0);
#else
    __builtin_amdgcn_tensor_load_to_lds(g0, g1, z4, z4, 0);
#endif
}
#endif

// ---------------------------------------------------------------------------
// Precompute DFT bases directly in WMMA B-fragment order.
// Fragment element (lane, h) of a 32x16 (KxN) B tile holds B[K][N] with
//   N = lane%16,  K = 16*(lane/16) + h.
// Forward basis bF: K=l (time), N=mc (col mc<64: cos(2pi*m*l/L); mc>=64: -sin)
// Inverse basis bI: K=kappa (0..63 real coeff, 64..127 imag coeff), N=l:
//   kappa<64 : ((kappa==0?1:2)/L) * cos(2pi*kappa*l/L)
//   kappa>=64: (-2/L) * sin(2pi*(kappa-64)*l/L)
// ---------------------------------------------------------------------------
__global__ __launch_bounds__(256) void init_basis(_Float16* __restrict__ bF,
                                                  _Float16* __restrict__ bI) {
    int idx = blockIdx.x * 256 + threadIdx.x;
    const float w0 = TWO_PI / (float)LSEQ;
    const int NF = LSEQ * K2;  // forward elements
    if (idx < NF) {
        int h    = idx & 15;
        int lane = (idx >> 4) & 31;
        int tile = idx >> 9;
        int nt = tile & 7;      // col tile (128 cols / 16)
        int kt = tile >> 3;     // k tile (4096 / 32)
        int l  = kt * 32 + ((lane >> 4) << 4) + h;
        int mc = nt * 16 + (lane & 15);
        int m  = mc & 63;
        int ml = (m * l) & (LSEQ - 1);
        float ang = (float)ml * w0;
        bF[idx] = (_Float16)((mc < 64) ? cosf(ang) : -sinf(ang));
    } else {
        int j = idx - NF;
        if (j < K2 * LSEQ) {
            int h    = j & 15;
            int lane = (j >> 4) & 31;
            int tile = j >> 9;
            int lt = tile & 255;   // col tile (4096 / 16)
            int kt = tile >> 8;    // k tile (128 / 32)
            int kk = kt * 32 + ((lane >> 4) << 4) + h;
            int l  = lt * 16 + (lane & 15);
            int m  = kk & 63;
            int ml = (m * l) & (LSEQ - 1);
            float ang = (float)ml * w0;
            float v;
            if (kk < 64) v = ((m == 0) ? 1.0f : 2.0f) * (1.0f / LSEQ) * cosf(ang);
            else         v = (-2.0f / LSEQ) * sinf(ang);
            bI[j] = (_Float16)v;
        }
    }
}

// ---------------------------------------------------------------------------
// Lift: x[b][h][l] = sum_c w_lift[h][c]*concat(u,z)[b][l][c] + b_lift[h], ->f16
// ---------------------------------------------------------------------------
__global__ __launch_bounds__(256) void lift_k(const float* __restrict__ u,
                                              const float* __restrict__ z,
                                              const float* __restrict__ wl,
                                              const float* __restrict__ bl,
                                              _Float16* __restrict__ x) {
    int idx = blockIdx.x * 256 + threadIdx.x;   // (b<<18)|(h<<12)|l
    int l = idx & (LSEQ - 1);
    int h = (idx >> 12) & 63;
    int b = idx >> 18;
    float acc = bl[h];
    const float* up = u + ((size_t)b * LSEQ + l) * 8;
    const float* wp = wl + h * 16;
    const float* zp = z + b * 8;
#pragma unroll
    for (int c = 0; c < 8; ++c) acc += wp[c] * up[c];
#pragma unroll
    for (int c = 0; c < 8; ++c) acc += wp[8 + c] * zp[c];
    x[idx] = (_Float16)acc;
}

// ---------------------------------------------------------------------------
// Forward truncated DFT GEMM: xf[B*H, 128] = x[B*H, L] @ basisF[L, 128]
// Workgroup: 16-row tile, 8 waves = the 8 column tiles. A tile staged in
// KCHUNK=256 chunks via double-buffered TDM: chunk i+1's DMA (16 x 512B,
// stride 8KB) is issued before computing chunk i, hidden behind 8 WMMAs.
// ---------------------------------------------------------------------------
__global__ __launch_bounds__(256) void fwd_dft(const _Float16* __restrict__ x,
                                               const _Float16* __restrict__ bF,
                                               float* __restrict__ xf) {
    __shared__ _Float16 lds_a[2][16 * KCHUNK];   // 16 KB double buffer
    int tid  = threadIdx.x;
    int wave = tid >> 5, lane = tid & 31;
    int r0 = blockIdx.x * 16;      // row tile in [0, B*H)
    int nt = wave;                 // col tile 0..7
    v8f c = {0.f, 0.f, 0.f, 0.f, 0.f, 0.f, 0.f, 0.f};
    int row16 = lane & 15;
    int kb    = (lane >> 4) * 16;  // A fragment K base: 0 or 16 (contiguous)
    const int NCH = LSEQ / KCHUNK; // 16 chunks

#if HAVE_TDM
    // D# group 1 (invariant): data_size=2B, tensor_dim0=tile_dim0=256,
    // tensor_dim1=tile_dim1=16, dim0 stride=4096 (data_size units).
    v8i_ g1;
    g1[0] = 0x00010000;
    g1[1] = (int)(256u << 16);
    g1[2] = (int)(16u << 16);
    g1[3] = (int)(256u << 16);
    g1[4] = 16;
    g1[5] = 4096;
    g1[6] = 0;
    g1[7] = 0;
    uint32_t lds0 = (uint32_t)(uintptr_t)(&lds_a[0][0]);
    uint32_t lds1 = (uint32_t)(uintptr_t)(&lds_a[1][0]);
    if (tid < 32)   // prologue: chunk 0 -> buffer 0
        tdm_issue_2d(lds0, x + (size_t)r0 * LSEQ, g1);
#endif

    for (int ci = 0; ci < NCH; ++ci) {
        __syncthreads();   // all waves done reading buffer (ci+1)&1
#if HAVE_TDM
        if (tid < 32) {
            if (ci + 1 < NCH) {   // prefetch next chunk into the other buffer
                tdm_issue_2d(((ci + 1) & 1) ? lds1 : lds0,
                             x + (size_t)r0 * LSEQ + (ci + 1) * KCHUNK, g1);
                __builtin_amdgcn_s_wait_tensorcnt(1);  // oldest (current) done
            } else {
                __builtin_amdgcn_s_wait_tensorcnt(0);
            }
        }
#else
        {   // fallback: cooperative vector staging (32 B per thread)
            int rr = tid >> 4, seg = tid & 15;
            const float4* s =
                (const float4*)(x + (size_t)(r0 + rr) * LSEQ + ci * KCHUNK + seg * 16);
            float4* d = (float4*)(&lds_a[ci & 1][0] + rr * KCHUNK + seg * 16);
            d[0] = s[0];
            d[1] = s[1];
        }
#endif
        __syncthreads();   // current buffer resident for all waves
        const _Float16* cur = &lds_a[ci & 1][0];
#pragma unroll
        for (int kt2 = 0; kt2 < KCHUNK / 32; ++kt2) {
            int kt = ci * (KCHUNK / 32) + kt2;
            Frag a;
            const float4* pa = (const float4*)(cur + row16 * KCHUNK + kt2 * 32 + kb);
            a.f4[0] = pa[0];   // K = kb .. kb+7
            a.f4[1] = pa[1];   // K = kb+8 .. kb+15
            v16h b = *(const v16h*)(bF + ((size_t)(kt * 8 + nt) * 32 + lane) * 16);
            c = __builtin_amdgcn_wmma_f32_16x16x32_f16(false, a.v, false, b,
                                                       (short)0, c, false, false);
        }
    }
    int col   = lane & 15;
    int rbase = (lane >> 4) * 8;
#pragma unroll
    for (int r = 0; r < 8; ++r)
        xf[(size_t)(r0 + rbase + r) * K2 + nt * 16 + col] = c[r];
}

// ---------------------------------------------------------------------------
// Per-mode complex mix: of[b,o,m] = sum_i xf[b,i,m] * (wr+j*wi)[o,i,m] -> f16
// ---------------------------------------------------------------------------
__global__ __launch_bounds__(256) void spec_mix(const float* __restrict__ xf,
                                                const float* __restrict__ wr,
                                                const float* __restrict__ wi,
                                                _Float16* __restrict__ of) {
    int idx = blockIdx.x * 256 + threadIdx.x;   // B*H*M
    int m = idx & 63;
    int o = (idx >> 6) & 63;
    int b = idx >> 12;
    const float* xb  = xf + (size_t)b * HCH * K2;
    const float* wrp = wr + (size_t)o * HCH * MMODE + m;
    const float* wip = wi + (size_t)o * HCH * MMODE + m;
    float ar = 0.f, ai = 0.f;
    for (int i = 0; i < HCH; ++i) {
        float xr = xb[i * K2 + m];
        float xi = xb[i * K2 + 64 + m];
        float wrv = wrp[i * MMODE], wiv = wip[i * MMODE];
        ar += xr * wrv - xi * wiv;
        ai += xr * wiv + xi * wrv;
    }
    of[(size_t)(b * HCH + o) * K2 + m]      = (_Float16)ar;
    of[(size_t)(b * HCH + o) * K2 + 64 + m] = (_Float16)ai;
}

// ---------------------------------------------------------------------------
// Fused iDFT + skip + bias + gelu (layers 0..2):
//   x'[b,o,l] = gelu( of[b,:,:] @ basisI[:,l]  +  w_skip[o,:] @ x[b,:,l] + bs[o] )
// Workgroup per (b, o-tile): stage of-tile (16x128) & w_skip tile (16x64) once.
// Each wave sweeps its share of the 256 l-tiles with PER-WAVE double-buffered
// TDM staging of the x tile (64 x 32B, stride 8KB) — no cross-wave barriers:
// issue tile idx+1, s_wait_tensorcnt(1), compute tile idx (4 iDFT + 2 skip
// WMMAs into one C fragment).
// ---------------------------------------------------------------------------
__global__ __launch_bounds__(256) void inv_skip(const _Float16* __restrict__ x,
                                                const _Float16* __restrict__ of,
                                                const _Float16* __restrict__ bI,
                                                const float* __restrict__ wsk,
                                                const float* __restrict__ bsk,
                                                _Float16* __restrict__ xo) {
    __shared__ _Float16 lds_of[16 * 128];
    __shared__ _Float16 lds_ws[16 * 64];
    __shared__ _Float16 lds_x[8][2][64 * 16];   // per-wave double buffer, 32 KB
    int tid = threadIdx.x, wave = tid >> 5, lane = tid & 31;
    int b  = blockIdx.x >> 2;
    int o0 = (blockIdx.x & 3) * 16;
    const _Float16* xb = x + (size_t)b * HCH * LSEQ;

    {   // stage of tile (2048 halfs) and w_skip tile (1024 f32 -> f16)
        int rr = tid >> 4, cc = (tid & 15) * 8;
        *(float4*)(lds_of + rr * 128 + cc) =
            *(const float4*)(of + (size_t)(b * HCH + o0 + rr) * K2 + cc);
        int i4 = tid * 4;
        float4 w = *(const float4*)(wsk + o0 * 64 + i4);
        _Float16* d = lds_ws + i4;
        d[0] = (_Float16)w.x; d[1] = (_Float16)w.y;
        d[2] = (_Float16)w.z; d[3] = (_Float16)w.w;
    }
    __syncthreads();

    int row16 = lane & 15;
    int kb    = (lane >> 4) * 16;
    Frag a_id[4];
#pragma unroll
    for (int kt = 0; kt < 4; ++kt) {
        const float4* p = (const float4*)(lds_of + row16 * 128 + kt * 32 + kb);
        a_id[kt].f4[0] = p[0]; a_id[kt].f4[1] = p[1];
    }
    Frag a_sk[2];
#pragma unroll
    for (int kt = 0; kt < 2; ++kt) {
        const float4* p = (const float4*)(lds_ws + row16 * 64 + kt * 32 + kb);
        a_sk[kt].f4[0] = p[0]; a_sk[kt].f4[1] = p[1];
    }
    int col = lane & 15, rbase = (lane >> 4) * 8;
    float bias[8];
#pragma unroll
    for (int r = 0; r < 8; ++r) bias[r] = bsk[o0 + rbase + r];

#if HAVE_TDM
    // D# group 1: data_size=2B, tensor_dim0=tile_dim0=16 halfs (32B rows),
    // tensor_dim1=tile_dim1=64, dim0 stride=4096.
    v8i_ g1x;
    g1x[0] = 0x00010000;
    g1x[1] = (int)(16u << 16);
    g1x[2] = (int)(64u << 16);
    g1x[3] = (int)(16u << 16);
    g1x[4] = 64;
    g1x[5] = 4096;
    g1x[6] = 0;
    g1x[7] = 0;
    uint32_t ldsx0 = (uint32_t)(uintptr_t)(&lds_x[wave][0][0]);
    uint32_t ldsx1 = (uint32_t)(uintptr_t)(&lds_x[wave][1][0]);
    tdm_issue_2d(ldsx0, xb + wave * 16, g1x);   // prologue: first tile
#endif

    int idx = 0;
    for (int lt = wave; lt < LSEQ / 16; lt += 8, ++idx) {
        int l0 = lt * 16;
        _Float16* myx = &lds_x[wave][idx & 1][0];
#if HAVE_TDM
        asm volatile("" ::: "memory");
        if (lt + 8 < LSEQ / 16) {   // prefetch this wave's next tile
            tdm_issue_2d((idx & 1) ? ldsx0 : ldsx1, xb + l0 + 128, g1x);
            __builtin_amdgcn_s_wait_tensorcnt(1);
        } else {
            __builtin_amdgcn_s_wait_tensorcnt(0);
        }
        asm volatile("" ::: "memory");
#else
        for (int ir = lane; ir < 64; ir += 32) {
            const float4* s = (const float4*)(xb + (size_t)ir * LSEQ + l0);
            *(float4*)(myx + ir * 16)     = s[0];
            *(float4*)(myx + ir * 16 + 8) = s[1];
        }
        __builtin_amdgcn_wave_barrier();
#endif

        v8f c = {0.f, 0.f, 0.f, 0.f, 0.f, 0.f, 0.f, 0.f};
#pragma unroll
        for (int kt = 0; kt < 4; ++kt) {
            v16h bb = *(const v16h*)(bI + ((size_t)(kt * 256 + lt) * 32 + lane) * 16);
            c = __builtin_amdgcn_wmma_f32_16x16x32_f16(false, a_id[kt].v, false, bb,
                                                       (short)0, c, false, false);
        }
#pragma unroll
        for (int kt = 0; kt < 2; ++kt) {
            Frag bb;
            int ib = kt * 32 + (lane >> 4) * 16;
#pragma unroll
            for (int h = 0; h < 16; ++h) bb.v[h] = myx[(ib + h) * 16 + col];
            c = __builtin_amdgcn_wmma_f32_16x16x32_f16(false, a_sk[kt].v, false, bb.v,
                                                       (short)0, c, false, false);
        }
#pragma unroll
        for (int r = 0; r < 8; ++r) {
            float v = gelu_tanh(c[r] + bias[r]);
            xo[(size_t)(b * HCH + o0 + rbase + r) * LSEQ + l0 + col] = (_Float16)v;
        }
#if !HAVE_TDM
        __builtin_amdgcn_wave_barrier();
#endif
    }
}

// ---------------------------------------------------------------------------
// Layer 3 (no gelu) evaluated only at l = L-1, fused with the projection.
// ---------------------------------------------------------------------------
__global__ __launch_bounds__(128) void final_k(const _Float16* __restrict__ x,
                                               const _Float16* __restrict__ of,
                                               const float* __restrict__ wsk3,
                                               const float* __restrict__ bsk3,
                                               const float* __restrict__ wp1,
                                               const float* __restrict__ bp1,
                                               const float* __restrict__ wp2,
                                               const float* __restrict__ bp2,
                                               float* __restrict__ out) {
    __shared__ float y[HCH];
    __shared__ float p1[2 * HCH];
    int b = blockIdx.x, tid = threadIdx.x;
    const float w0 = TWO_PI / (float)LSEQ;
    if (tid < HCH) {
        int o = tid;
        float xs = 0.f;
        for (int k = 0; k < K2; ++k) {
            int m  = k & 63;
            int ml = (m * (LSEQ - 1)) & (LSEQ - 1);
            float ang = (float)ml * w0;
            float basis = (k < 64)
                ? ((m == 0) ? 1.0f : 2.0f) * (1.0f / LSEQ) * cosf(ang)
                : (-2.0f / LSEQ) * sinf(ang);
            xs += (float)of[(size_t)(b * HCH + o) * K2 + k] * basis;
        }
        float sk = bsk3[o];
        for (int i = 0; i < HCH; ++i)
            sk += wsk3[o * HCH + i] * (float)x[(size_t)(b * HCH + i) * LSEQ + (LSEQ - 1)];
        y[o] = xs + sk;   // no gelu after last layer
    }
    __syncthreads();
    {
        float acc = bp1[tid];
        for (int o = 0; o < HCH; ++o) acc += wp1[tid * HCH + o] * y[o];
        p1[tid] = gelu_tanh(acc);
    }
    __syncthreads();
    if (tid < 8) {
        float acc = bp2[tid];
        for (int p = 0; p < 2 * HCH; ++p) acc += wp2[tid * 2 * HCH + p] * p1[p];
        out[b * 8 + tid] = acc;
    }
}

// ---------------------------------------------------------------------------
extern "C" void kernel_launch(void* const* d_in, const int* in_sizes, int n_in,
                              void* d_out, int out_size, void* d_ws, size_t ws_size,
                              hipStream_t stream) {
    const float* u       = (const float*)d_in[0];
    const float* z       = (const float*)d_in[1];
    // d_in[2] = t (unused by reference)
    const float* w_lift  = (const float*)d_in[3];
    const float* b_lift  = (const float*)d_in[4];
    const float* spec_wr = (const float*)d_in[5];
    const float* spec_wi = (const float*)d_in[6];
    const float* w_skip  = (const float*)d_in[7];
    const float* b_skip  = (const float*)d_in[8];
    const float* w_p1    = (const float*)d_in[9];
    const float* b_p1    = (const float*)d_in[10];
    const float* w_p2    = (const float*)d_in[11];
    const float* b_p2    = (const float*)d_in[12];
    float* out = (float*)d_out;

    // workspace carve-up (~35.5 MB total)
    char* ws = (char*)d_ws;
    size_t off = 0;
    auto carve = [&](size_t bytes) -> void* {
        void* p = ws + off;
        off += (bytes + 255) & ~(size_t)255;
        return p;
    };
    _Float16* bF = (_Float16*)carve((size_t)LSEQ * K2 * 2);           // 1 MB
    _Float16* bI = (_Float16*)carve((size_t)K2 * LSEQ * 2);           // 1 MB
    _Float16* xA = (_Float16*)carve((size_t)BATCH * HCH * LSEQ * 2);  // 16 MB
    _Float16* xB = (_Float16*)carve((size_t)BATCH * HCH * LSEQ * 2);  // 16 MB
    float*    xf = (float*)   carve((size_t)BATCH * HCH * K2 * 4);    // 1 MB
    _Float16* of = (_Float16*)carve((size_t)BATCH * HCH * K2 * 2);    // 0.5 MB
    (void)in_sizes; (void)n_in; (void)out_size; (void)ws_size;

    init_basis<<<(2 * LSEQ * K2) / 256, 256, 0, stream>>>(bF, bI);
    lift_k<<<(BATCH * HCH * LSEQ) / 256, 256, 0, stream>>>(u, z, w_lift, b_lift, xA);

    _Float16* xc = xA;
    _Float16* xn = xB;
    for (int layer = 0; layer < 4; ++layer) {
        fwd_dft<<<BATCH * HCH / 16, 256, 0, stream>>>(xc, bF, xf);
        spec_mix<<<(BATCH * HCH * MMODE) / 256, 256, 0, stream>>>(
            xf, spec_wr + (size_t)layer * HCH * HCH * MMODE,
                spec_wi + (size_t)layer * HCH * HCH * MMODE, of);
        if (layer < 3) {
            inv_skip<<<BATCH * 4, 256, 0, stream>>>(
                xc, of, bI, w_skip + (size_t)layer * HCH * HCH,
                b_skip + (size_t)layer * HCH, xn);
            _Float16* t = xc; xc = xn; xn = t;
        } else {
            final_k<<<BATCH, 128, 0, stream>>>(
                xc, of, w_skip + (size_t)3 * HCH * HCH, b_skip + (size_t)3 * HCH,
                w_p1, b_p1, w_p2, b_p2, out);
        }
    }
}
